// LSTMModel_6279242187438
// MI455X (gfx1250) — matrix operand order repaired and checked by reading, not compiled, not run on hardware
//
#include <hip/hip_runtime.h>
#include <hip/hip_bf16.h>

typedef __attribute__((ext_vector_type(16))) __bf16 v16bf;
typedef __attribute__((ext_vector_type(8)))  __bf16 v8bf;
typedef __attribute__((ext_vector_type(8)))  float  v8f;
typedef __attribute__((ext_vector_type(4)))  float  v4f;

#define Bn   512
#define Tn   512
#define Fn   16
#define Hn   256
#define GN   1024   /* 4*H */
#define HORn 24

// Packed B-fragment regions (bf16 elements):
//  pack[((nt*ktc + kt)*32 + lane)*16 + j] = W[N][K]
//  with N = nt*16 + (lane&15), K = kt*32 + (lane>=16 ? 8 : 0) + (j&7) + (j>=8 ? 16 : 0)
static constexpr int SZ_IH0 = 64 * 1 * 32 * 16;   // W_ih0, K=16 zero-padded to 32
static constexpr int SZ_M   = 64 * 8 * 32 * 16;   // K=256 matrices
static constexpr int PACK_TOTAL = SZ_IH0 + 5 * SZ_M;

__global__ void pack_weights(const float* __restrict__ W_ih0,
                             const float* __restrict__ W_ih12,
                             const float* __restrict__ W_hh,
                             __bf16* __restrict__ pack)
{
  int idx = blockIdx.x * blockDim.x + threadIdx.x;
  if (idx >= PACK_TOTAL) return;
  const float* W; int ktc, Keff, local;
  if (idx < SZ_IH0) {
    W = W_ih0; ktc = 1; Keff = Fn; local = idx;
  } else if (idx < SZ_IH0 + 2 * SZ_M) {
    int r = (idx - SZ_IH0) / SZ_M;
    W = W_ih12 + (size_t)r * GN * Hn; ktc = 8; Keff = Hn;
    local = (idx - SZ_IH0) % SZ_M;
  } else {
    int r = (idx - SZ_IH0 - 2 * SZ_M) / SZ_M;
    W = W_hh + (size_t)r * GN * Hn; ktc = 8; Keff = Hn;
    local = (idx - SZ_IH0 - 2 * SZ_M) % SZ_M;
  }
  int j    = local & 15;
  int lane = (local >> 4) & 31;
  int kt   = (local >> 9) % ktc;
  int nt   = (local >> 9) / ktc;
  int N = nt * 16 + (lane & 15);
  int K = kt * 32 + ((lane & 16) ? 8 : 0) + (j & 7) + ((j & 8) ? 16 : 0);
  float v = (K < Keff) ? W[(size_t)N * Keff + K] : 0.0f;
  pack[idx] = (__bf16)v;
}

__device__ __forceinline__ v16bf cat8(v8bf lo, v8bf hi) {
  return __builtin_shufflevector(lo, hi, 0,1,2,3,4,5,6,7,8,9,10,11,12,13,14,15);
}

template<bool REUSE_A>
__device__ __forceinline__ v8f wmma_bf16(v16bf a, v16bf b, v8f c) {
  return __builtin_amdgcn_wmma_f32_16x16x32_bf16(
      /*neg_a=*/false, a, /*neg_b=*/false, b,
      /*c_mod=*/(short)0, c, /*reuse_a=*/REUSE_A, /*reuse_b=*/false);
}

// A fragment (16xK row-major, bf16) from LDS, ISA A-layout:
// lane<16: row=lane,    K = kt*32 + {0..7, 16..23}
// lane>=16: row=lane-16, K = kt*32 + {8..15, 24..31}
__device__ __forceinline__ v16bf loadA(const __bf16 (*h)[264], int kt, int lane) {
  int row   = lane & 15;
  int kbase = kt * 32 + ((lane & 16) ? 8 : 0);
  v8bf lo = *(const v8bf*)&h[row][kbase];
  v8bf hi = *(const v8bf*)&h[row][kbase + 16];
  return cat8(lo, hi);
}

// B fragment from a packed weight region: 32 contiguous bytes per lane,
// 32-bit lane offset from a uniform (SGPR) base -> global_load saddr form.
__device__ __forceinline__ v16bf loadB(const __bf16* __restrict__ pk,
                                       int nt, int kt, int ktc, int lane) {
  unsigned off = (unsigned)((((nt * ktc + kt) << 5) + lane) << 4);
  const __bf16* p = pk + off;
  v8bf lo = *(const v8bf*)p;
  v8bf hi = *(const v8bf*)(p + 8);
  return cat8(lo, hi);
}

__device__ __forceinline__ float sigm(float x) {
  return 1.0f / (1.0f + __expf(-x));
}
__device__ __forceinline__ float tanh_f(float x) {
  x = fminf(fmaxf(x, -15.0f), 15.0f);
  float e = __expf(-2.0f * x);
  return (1.0f - e) / (1.0f + e);
}

// One K-step: 8 WMMAs sharing the A fragment; first issues without the
// reuse hint, the remaining 7 set reuse_a (identical instruction form).
#define WMMA_GROUP(pk, kt, ktc)                                             \
  {                                                                         \
    { v16bf bm = loadB(pk, tiles[0], kt, ktc, lane);                        \
      acc[0] = wmma_bf16<false>(a, bm, acc[0]); }                           \
    { v16bf bm = loadB(pk, tiles[1], kt, ktc, lane);                        \
      acc[1] = wmma_bf16<true>(a, bm, acc[1]); }                            \
    { v16bf bm = loadB(pk, tiles[2], kt, ktc, lane);                        \
      acc[2] = wmma_bf16<true>(a, bm, acc[2]); }                            \
    { v16bf bm = loadB(pk, tiles[3], kt, ktc, lane);                        \
      acc[3] = wmma_bf16<true>(a, bm, acc[3]); }                            \
    { v16bf bm = loadB(pk, tiles[4], kt, ktc, lane);                        \
      acc[4] = wmma_bf16<true>(a, bm, acc[4]); }                            \
    { v16bf bm = loadB(pk, tiles[5], kt, ktc, lane);                        \
      acc[5] = wmma_bf16<true>(a, bm, acc[5]); }                            \
    { v16bf bm = loadB(pk, tiles[6], kt, ktc, lane);                        \
      acc[6] = wmma_bf16<true>(a, bm, acc[6]); }                            \
    { v16bf bm = loadB(pk, tiles[7], kt, ktc, lane);                        \
      acc[7] = wmma_bf16<true>(a, bm, acc[7]); }                            \
  }

// 16x1024 += (16x256 from LDS) x packed-W, K = 8 tiles of 32
__device__ __forceinline__ void gemm8(v8f acc[8], const __bf16 (*h)[264],
                                      const __bf16* __restrict__ pk,
                                      const int tiles[8], int lane) {
#pragma unroll 1
  for (int kt = 0; kt < 8; kt++) {
    v16bf a = loadA(h, kt, lane);
    WMMA_GROUP(pk, kt, 8)
  }
}

// LSTM pointwise on this wave's 2 hidden tiles; writes new h into LDS
__device__ __forceinline__ void pointwise(v8f acc[8], float cr[2][8],
                                          __bf16 (*hself)[264],
                                          int wave, int lane) {
  const int lrow = lane & 15;
  const int mofs = (lane & 16) ? 8 : 0;
#pragma unroll
  for (int ht = 0; ht < 2; ht++) {
#pragma unroll
    for (int r = 0; r < 8; r++) {
      float ig = sigm(acc[0 + ht][r]);
      float fg = sigm(acc[2 + ht][r]);
      float gg = tanh_f(acc[4 + ht][r]);
      float og = sigm(acc[6 + ht][r]);
      float cc = fg * cr[ht][r] + ig * gg;
      cr[ht][r] = cc;
      float hh = og * tanh_f(cc);
      hself[r + mofs][(2 * wave + ht) * 16 + lrow] = (__bf16)hh;
    }
  }
}

__global__ __launch_bounds__(256, 1)
void lstm_fused(const float* __restrict__ x,
                const float* __restrict__ b_ih, const float* __restrict__ b_hh,
                const float* __restrict__ fc_W, const float* __restrict__ fc_b,
                const __bf16* __restrict__ pack,
                float* __restrict__ out)
{
  __shared__ __align__(16) __bf16 hbuf[3][16][264];  // 528B row stride (bank-safe)

  const int lane = threadIdx.x & 31;
  const int wave = threadIdx.x >> 5;        // 0..7
  const int b0   = blockIdx.x * 16;         // batch tile base
  const int lrow = lane & 15;
  const int hi16 = (lane & 16) ? 1 : 0;

  for (int i = threadIdx.x; i < 3 * 16 * 264; i += 256)
    ((__bf16*)hbuf)[i] = (__bf16)0.0f;

  // gate N-tiles owned by this wave: i0,i1,f0,f1,g0,g1,o0,o1
  int tiles[8];
#pragma unroll
  for (int j = 0; j < 8; j++) tiles[j] = 2 * wave + (j & 1) + (j >> 1) * 16;

  // fused bias (constant along M => broadcast across all 8 acc VGPRs)
  float bias[3][8];
#pragma unroll
  for (int l = 0; l < 3; l++)
#pragma unroll
    for (int j = 0; j < 8; j++) {
      int N = tiles[j] * 16 + lrow;
      bias[l][j] = b_ih[l * GN + N] + b_hh[l * GN + N];
    }

  // cell state in registers, accumulator layout
  float creg[3][2][8];
#pragma unroll
  for (int l = 0; l < 3; l++)
#pragma unroll
    for (int h = 0; h < 2; h++)
#pragma unroll
      for (int r = 0; r < 8; r++) creg[l][h][r] = 0.0f;

  // uniform SGPR region bases (layer stages fully unrolled below)
  const __bf16* pk_ih0 = pack;
  const __bf16* pk_ih1 = pack + SZ_IH0;
  const __bf16* pk_ih2 = pack + SZ_IH0 + SZ_M;
  const __bf16* pk_hh0 = pack + SZ_IH0 + 2 * SZ_M;
  const __bf16* pk_hh1 = pack + SZ_IH0 + 3 * SZ_M;
  const __bf16* pk_hh2 = pack + SZ_IH0 + 4 * SZ_M;

  __syncthreads();

#pragma unroll 1
  for (int t = 0; t < Tn; t++) {
    v8f acc[8];

    // ================= layer 0 =================
#pragma unroll
    for (int j = 0; j < 8; j++)
#pragma unroll
      for (int r = 0; r < 8; r++) acc[j][r] = bias[0][j];
    {
      // A fragment straight from x (K=16, upper half zero-padded)
      const float* xp = x + (((size_t)(b0 + lrow)) * Tn + t) * Fn + hi16 * 8;
      v4f x0 = *(const v4f*)xp;
      v4f x1 = *(const v4f*)(xp + 4);
      v16bf a;
#pragma unroll
      for (int j = 0; j < 4; j++) { a[j] = (__bf16)x0[j]; a[4 + j] = (__bf16)x1[j]; }
#pragma unroll
      for (int j = 8; j < 16; j++) a[j] = (__bf16)0.0f;
      WMMA_GROUP(pk_ih0, 0, 1)
    }
    gemm8(acc, hbuf[0], pk_hh0, tiles, lane);     // recurrent
    __syncthreads();                              // readers of old h0 done
    pointwise(acc, creg[0], hbuf[0], wave, lane);
    __syncthreads();                              // new h0 published

    // ================= layer 1 =================
#pragma unroll
    for (int j = 0; j < 8; j++)
#pragma unroll
      for (int r = 0; r < 8; r++) acc[j][r] = bias[1][j];
    gemm8(acc, hbuf[0], pk_ih1, tiles, lane);     // input  (h0 new)
    gemm8(acc, hbuf[1], pk_hh1, tiles, lane);     // recurrent (h1 old)
    __syncthreads();
    pointwise(acc, creg[1], hbuf[1], wave, lane);
    __syncthreads();

    // ================= layer 2 =================
#pragma unroll
    for (int j = 0; j < 8; j++)
#pragma unroll
      for (int r = 0; r < 8; r++) acc[j][r] = bias[2][j];
    gemm8(acc, hbuf[1], pk_ih2, tiles, lane);     // input  (h1 new)
    gemm8(acc, hbuf[2], pk_hh2, tiles, lane);     // recurrent (h2 old)
    __syncthreads();
    pointwise(acc, creg[2], hbuf[2], wave, lane);
    __syncthreads();
  }

  // ---- FC head on last hidden of layer 2 ----
  for (int task = threadIdx.x; task < 16 * HORn; task += 256) {
    int row = task / HORn, ho = task % HORn;
    float s = fc_b[ho];
    for (int k = 0; k < Hn; k++)
      s += (float)hbuf[2][row][k] * fc_W[ho * Hn + k];
    out[(size_t)(b0 + row) * HORn + ho] = s;
  }
}

extern "C" void kernel_launch(void* const* d_in, const int* in_sizes, int n_in,
                              void* d_out, int out_size, void* d_ws, size_t ws_size,
                              hipStream_t stream)
{
  const float* x      = (const float*)d_in[0];
  const float* W_ih0  = (const float*)d_in[1];
  const float* W_ih12 = (const float*)d_in[2];
  const float* W_hh   = (const float*)d_in[3];
  const float* b_ih   = (const float*)d_in[4];
  const float* b_hh   = (const float*)d_in[5];
  const float* fc_W   = (const float*)d_in[6];
  const float* fc_b   = (const float*)d_in[7];
  float* outp  = (float*)d_out;
  __bf16* pack = (__bf16*)d_ws;   // 2.69 MB of ws used

  pack_weights<<<(PACK_TOTAL + 255) / 256, 256, 0, stream>>>(W_ih0, W_ih12, W_hh, pack);
  lstm_fused<<<Bn / 16, 256, 0, stream>>>(x, b_ih, b_hh, fc_W, fc_b, pack, outp);
}
